// Vit_1219770712350
// MI455X (gfx1250) — compile-verified, matrix operand order checked by reading, not run on hardware
//
#include <hip/hip_runtime.h>
#include <stdint.h>

// ---------------------------------------------------------------------------
// ViT attention, cls-token-only output, algebraically pruned (see analysis).
// MI455X / gfx1250: wave32, WMMA bf16 (f32 accumulate).
// ---------------------------------------------------------------------------

typedef __attribute__((ext_vector_type(16))) __bf16 v16bf;
typedef __attribute__((ext_vector_type(8)))  float  v8f;

#define DIMC   2048
#define SEQC   512
#define NB     16
#define NHEAD  16
#define HD     128
#define QKVLD  6144   // qkv_w row stride

// ---- WMMA fragment gather: 16-bit A/B layout (ISA 7.12.2) -----------------
// lane l in [0,16): M(or N)=l ; hi=lane>>4 selects K+8 ; VGPR j pairs:
//   j<4 : K = 2j + 8*hi ; j>=4 : K = 16 + 2(j-4) + 8*hi
__device__ __forceinline__ v16bf frag_from_row(const __bf16* row, int k0, int hi) {
  v16bf f;
#pragma unroll
  for (int j = 0; j < 8; ++j) {
    int kb = 2 * j + ((j & 4) ? 8 : 0) + 8 * hi;
    uint32_t w;
    __builtin_memcpy(&w, row + k0 + kb, 4);
    union { uint32_t u; __bf16 h[2]; } cv; cv.u = w;
    f[2 * j]     = cv.h[0];
    f[2 * j + 1] = cv.h[1];
  }
  return f;
}

// ---- Stage 0: tok = concat(cls, permute(x)) + pos  -> bf16 ----------------
__global__ void vit_build_tok(const float* __restrict__ x,
                              const float* __restrict__ pos,
                              const float* __restrict__ cls,
                              __bf16* __restrict__ tok) {
  size_t pid = (size_t)blockIdx.x * 256 + threadIdx.x;  // one bf16x2 per thread
  size_t e2  = pid * 2;
  int e  = (int)(e2 & (DIMC - 1));
  size_t bs = e2 >> 11;
  int s  = (int)(bs & (SEQC - 1));
  int b  = (int)(bs >> 9);
  float f0, f1;
  if (s == 0) {
    f0 = cls[e]; f1 = cls[e + 1];
  } else {
    int c = e >> 8, r = e & 255;                 // e = c*256 + (h*16+w)
    const float* px = x + (((size_t)(b * 8 + c) * 511 + (s - 1)) * 256 + r);
    f0 = px[0]; f1 = px[1];
  }
  f0 += pos[(size_t)s * DIMC + e];
  f1 += pos[(size_t)s * DIMC + e + 1];
  union { uint32_t u; __bf16 h[2]; } cv;
  cv.h[0] = (__bf16)f0; cv.h[1] = (__bf16)f1;
  *(uint32_t*)(tok + e2) = cv.u;
}

// ---- Stage 1: q = (cls+pos0) @ Wq * scale  (batch-independent) ------------
__global__ void vit_q(const float* __restrict__ cls, const float* __restrict__ pos,
                      const float* __restrict__ qkv_w, const float* __restrict__ qkv_b,
                      float* __restrict__ qs) {
  __shared__ float tok0[DIMC];
  int t = threadIdx.x;                       // 128 threads
  for (int i = t; i < DIMC; i += 128) tok0[i] = cls[i] + pos[i];
  __syncthreads();
  int col = blockIdx.x * 128 + t;
  float acc = 0.f;
  for (int e = 0; e < DIMC; ++e)
    acc = fmaf(tok0[e], qkv_w[(size_t)e * QKVLD + col], acc);
  qs[col] = (acc + qkv_b[col]) * 0.08838834764831845f;   // 1/sqrt(128)
}

// ---- Stage 1b: w~T[h][e] = sum_d Wk[e, h*128+d] * qs[h*128+d]  -> bf16 ----
__global__ void vit_wtilde(const float* __restrict__ qkv_w,
                           const float* __restrict__ qs,
                           __bf16* __restrict__ wT) {
  __shared__ float prod[DIMC];
  int e = blockIdx.x, t = threadIdx.x;       // 128 threads
  const float* row = qkv_w + (size_t)e * QKVLD + DIMC;   // K-projection slice
  for (int j = t; j < DIMC; j += 128) prod[j] = row[j] * qs[j];
  __syncthreads();
  if (t < NHEAD) {
    float s = 0.f;
    for (int d = 0; d < HD; ++d) s += prod[t * HD + d];
    wT[(size_t)t * DIMC + e] = (__bf16)s;
  }
}

__global__ void vit_cbias(const float* __restrict__ qs, const float* __restrict__ qkv_b,
                          float* __restrict__ cb) {
  int h = threadIdx.x;                       // 16 threads
  float s = 0.f;
  for (int d = 0; d < HD; ++d) s += qs[h * HD + d] * qkv_b[DIMC + h * HD + d];
  cb[h] = s;
}

// ---- Stage 2: pre-softmax scores via WMMA: S[b,h,s] = tok @ w~T + c[h] ----
__global__ void vit_scores_wmma(const __bf16* __restrict__ tok,
                                const __bf16* __restrict__ wT,
                                const float* __restrict__ cbias,
                                float* __restrict__ S) {
  int wid  = blockIdx.x * (blockDim.x >> 5) + (threadIdx.x >> 5);
  int b    = wid >> 5;                       // 32 s-tiles per batch
  int st   = wid & 31;
  int lane = threadIdx.x & 31, l = lane & 15, hi = lane >> 4;
  const __bf16* rowA = tok + (size_t)(b * SEQC + st * 16 + l) * DIMC;  // A: 16 s-rows
  const __bf16* rowB = wT  + (size_t)l * DIMC;                         // B: 16 heads
  v8f c = {};
#pragma unroll 4
  for (int k0 = 0; k0 < DIMC; k0 += 32) {
    v16bf a  = frag_from_row(rowA, k0, hi);
    v16bf bf = frag_from_row(rowB, k0, hi);
    c = __builtin_amdgcn_wmma_f32_16x16x32_bf16(false, a, false, bf, (short)0,
                                                c, false, false);
  }
  float cbv = cbias[l];                      // D: n = head = l, m = r + 8*hi = s
  float* out = S + ((size_t)(b * NHEAD + l)) * SEQC + st * 16 + 8 * hi;
#pragma unroll
  for (int r = 0; r < 8; ++r) out[r] = c[r] + cbv;
}

// ---- Stage 3: softmax over s (512) per (b,h), store bf16 ------------------
__global__ void vit_softmax(const float* __restrict__ S, __bf16* __restrict__ P) {
  __shared__ float red[256];
  int bh = blockIdx.x, t = threadIdx.x;      // 256 threads, 2 values each
  const float* row = S + (size_t)bh * SEQC;
  float v0 = row[t], v1 = row[t + 256];
  red[t] = fmaxf(v0, v1); __syncthreads();
  for (int s = 128; s > 0; s >>= 1) { if (t < s) red[t] = fmaxf(red[t], red[t + s]); __syncthreads(); }
  float m = red[0]; __syncthreads();
  float e0 = __expf(v0 - m), e1 = __expf(v1 - m);
  red[t] = e0 + e1; __syncthreads();
  for (int s = 128; s > 0; s >>= 1) { if (t < s) red[t] += red[t + s]; __syncthreads(); }
  float inv = 1.0f / red[0];
  __bf16* pr = P + (size_t)bh * SEQC;
  pr[t]       = (__bf16)(e0 * inv);
  pr[t + 256] = (__bf16)(e1 * inv);
}

// ---- Stage 4: T[b,h,e] = p(16x512) @ tok(512x2048) via WMMA ---------------
// tok tile staged transposed through LDS so B-fragment gathers are dword reads.
__global__ void vit_attnv_wmma(const __bf16* __restrict__ tok,
                               const __bf16* __restrict__ P,
                               float* __restrict__ T) {
  __shared__ __bf16 bt[128][36];             // [e_local][k], pad for alignment
  int b   = blockIdx.x >> 4;
  int e0b = (blockIdx.x & 15) * 128;
  int wv  = threadIdx.x >> 5;                // 8 waves -> 8 N-tiles of 16
  int lane = threadIdx.x & 31, l = lane & 15, hi = lane >> 4;
  const __bf16* rowA = P + (size_t)(b * NHEAD + l) * SEQC;
  v8f c = {};
  for (int k0 = 0; k0 < SEQC; k0 += 32) {
    __syncthreads();
    for (int idx = threadIdx.x; idx < 2048; idx += 256) {   // 32k x 128e bf16 tile
      int kk = idx >> 6;                     // 64 dwords per k-row
      int ep = (idx & 63) * 2;
      uint32_t w;
      __builtin_memcpy(&w, tok + (size_t)(b * SEQC + k0 + kk) * DIMC + e0b + ep, 4);
      union { uint32_t u; __bf16 h[2]; } cv; cv.u = w;
      bt[ep][kk]     = cv.h[0];
      bt[ep + 1][kk] = cv.h[1];
    }
    __syncthreads();
    v16bf a = frag_from_row(rowA, k0, hi);
    v16bf bf;
    const __bf16* brow = &bt[wv * 16 + l][0];
#pragma unroll
    for (int j = 0; j < 8; ++j) {
      int kb = 2 * j + ((j & 4) ? 8 : 0) + 8 * hi;
      uint32_t w; __builtin_memcpy(&w, brow + kb, 4);
      union { uint32_t u; __bf16 h[2]; } cv; cv.u = w;
      bf[2 * j] = cv.h[0]; bf[2 * j + 1] = cv.h[1];
    }
    c = __builtin_amdgcn_wmma_f32_16x16x32_bf16(false, a, false, bf, (short)0,
                                                c, false, false);
  }
  float* out = T + (size_t)b * NHEAD * DIMC;
  int e = e0b + wv * 16 + l;
#pragma unroll
  for (int r = 0; r < 8; ++r)
    out[(size_t)(r + 8 * hi) * DIMC + e] = c[r];
}

// ---- Stage 5: ctx[b, h*128+d] = T[b,h,:] @ Wv[:, 4096+h*128+d] + bv -------
__global__ void vit_ctx(const float* __restrict__ T, const float* __restrict__ qkv_w,
                        const float* __restrict__ qkv_b, float* __restrict__ ctx) {
  __shared__ float tl[NB][256];
  int h = blockIdx.x, t = threadIdx.x;       // 128 threads = 128 d's
  float acc[NB];
#pragma unroll
  for (int b = 0; b < NB; ++b) acc[b] = 0.f;
  int col = 2 * DIMC + h * HD + t;           // V slice of qkv_w
  for (int ec = 0; ec < DIMC; ec += 256) {
    __syncthreads();
    for (int i = t; i < NB * 256; i += 128) {
      int b = i >> 8, e = i & 255;
      tl[b][e] = T[(size_t)(b * NHEAD + h) * DIMC + ec + e];
    }
    __syncthreads();
    for (int e = 0; e < 256; ++e) {
      float w = qkv_w[(size_t)(ec + e) * QKVLD + col];
#pragma unroll
      for (int b = 0; b < NB; ++b) acc[b] = fmaf(tl[b][e], w, acc[b]);
    }
  }
  float bias = qkv_b[col];
  for (int b = 0; b < NB; ++b) ctx[(size_t)b * DIMC + h * HD + t] = acc[b] + bias;
}

// ---- Stage 6: out = ctx @ proj_w + proj_b ---------------------------------
__global__ void vit_out(const float* __restrict__ ctx, const float* __restrict__ proj_w,
                        const float* __restrict__ proj_b, float* __restrict__ out) {
  __shared__ float cl[NB][256];
  int t = threadIdx.x;                       // 128 threads
  int col = blockIdx.x * 128 + t;
  float acc[NB];
#pragma unroll
  for (int b = 0; b < NB; ++b) acc[b] = 0.f;
  for (int ec = 0; ec < DIMC; ec += 256) {
    __syncthreads();
    for (int i = t; i < NB * 256; i += 128) {
      int b = i >> 8, e = i & 255;
      cl[b][e] = ctx[(size_t)b * DIMC + ec + e];
    }
    __syncthreads();
    for (int e = 0; e < 256; ++e) {
      float w = proj_w[(size_t)(ec + e) * DIMC + col];
#pragma unroll
      for (int b = 0; b < NB; ++b) acc[b] = fmaf(cl[b][e], w, acc[b]);
    }
  }
  float bias = proj_b[col];
  for (int b = 0; b < NB; ++b) out[(size_t)b * DIMC + col] = acc[b] + bias;
}

// ---------------------------------------------------------------------------
extern "C" void kernel_launch(void* const* d_in, const int* in_sizes, int n_in,
                              void* d_out, int out_size, void* d_ws, size_t ws_size,
                              hipStream_t stream) {
  const float* x      = (const float*)d_in[0];
  const float* pos    = (const float*)d_in[1];
  const float* cls    = (const float*)d_in[2];
  const float* qkv_w  = (const float*)d_in[3];
  const float* qkv_b  = (const float*)d_in[4];
  const float* proj_w = (const float*)d_in[5];
  const float* proj_b = (const float*)d_in[6];
  float* out = (float*)d_out;

  char* ws = (char*)d_ws;
  size_t off = 0;
  auto take = [&](size_t bytes) { void* p = ws + off; off += (bytes + 255) & ~(size_t)255; return p; };
  __bf16* tok = (__bf16*)take((size_t)NB * SEQC * DIMC * 2);   // 32 MB
  float*  S   = (float*) take((size_t)NB * NHEAD * SEQC * 4);  // 512 KB
  __bf16* P   = (__bf16*)take((size_t)NB * NHEAD * SEQC * 2);  // 256 KB
  float*  T   = (float*) take((size_t)NB * NHEAD * DIMC * 4);  // 2 MB
  float*  qs  = (float*) take((size_t)DIMC * 4);
  __bf16* wT  = (__bf16*)take((size_t)NHEAD * DIMC * 2);
  float*  cb  = (float*) take((size_t)NHEAD * 4);
  float*  ctx = (float*) take((size_t)NB * DIMC * 4);

  vit_build_tok <<<(NB * SEQC * DIMC) / 512, 256, 0, stream>>>(x, pos, cls, tok);
  vit_q         <<<DIMC / 128, 128, 0, stream>>>(cls, pos, qkv_w, qkv_b, qs);
  vit_wtilde    <<<DIMC, 128, 0, stream>>>(qkv_w, qs, wT);
  vit_cbias     <<<1, NHEAD, 0, stream>>>(qs, qkv_b, cb);
  vit_scores_wmma<<<(NB * 32) / 8, 256, 0, stream>>>(tok, wT, cb, S);
  vit_softmax   <<<NB * NHEAD, 256, 0, stream>>>(S, P);
  vit_attnv_wmma<<<NB * (DIMC / 128), 256, 0, stream>>>(tok, P, T);
  vit_ctx       <<<NHEAD, 128, 0, stream>>>(T, qkv_w, qkv_b, ctx);
  vit_out       <<<DIMC / 128, 128, 0, stream>>>(ctx, proj_w, proj_b, out);
}
